// RNN_69441031241740
// MI455X (gfx1250) — matrix operand order repaired and checked by reading, not compiled, not run on hardware
//
#include <hip/hip_runtime.h>
#include <hip/hip_bf16.h>

#define B_    64
#define S_    512
#define DIN_  512
#define H_    1024
#define DOUT_ 1024
#define MS_   (B_ * S_)   // 32768 rows for the big GEMMs

#define STEP_BLOCKS 32    // persistent kernel grid (16 layer0 + 16 layer1)

typedef __attribute__((ext_vector_type(16))) _Float16 v16h;
typedef __attribute__((ext_vector_type(8)))  _Float16 v8h;
typedef __attribute__((ext_vector_type(8)))  float    v8f;

__device__ __forceinline__ v8f wmma16(v16h a, v16h b, v8f c) {
  // D = A(16x32 f16) * B(32x16 f16) + C(16x16 f32)
  return __builtin_amdgcn_wmma_f32_16x16x32_f16(
      false, a, false, b, (short)0, c, false, false);
}

// ---- Fragment loaders (ISA 7.12.2 layouts, wave32) ----

// A fragment (16x32 f16), row-major source, leading dim lda.
__device__ __forceinline__ v16h load_a_h(const _Float16* A, int lda, int m0, int k0, int lane) {
  const int m  = m0 + (lane & 15);
  const int hh = (lane >> 4) * 8;
  const _Float16* p = A + (size_t)m * lda + k0 + hh;
  v8h lo = *(const v8h*)p;
  v8h hi = *(const v8h*)(p + 16);
  v16h r;
#pragma unroll
  for (int i = 0; i < 8; ++i) { r[i] = lo[i]; r[i + 8] = hi[i]; }
  return r;
}

// A fragment from f32 source (convert on the fly).
__device__ __forceinline__ v16h load_a_f(const float* A, int lda, int m0, int k0, int lane) {
  const int m  = m0 + (lane & 15);
  const int hh = (lane >> 4) * 8;
  const float* p = A + (size_t)m * lda + k0 + hh;
  v16h r;
#pragma unroll
  for (int i = 0; i < 8; ++i) { r[i] = (_Float16)p[i]; r[i + 8] = (_Float16)p[i + 16]; }
  return r;
}

// B fragment (32x16): B[k][n] = W[n][k], W row-major [N][ldw].
__device__ __forceinline__ v16h load_b_w(const _Float16* W, int ldw, int n0, int k0, int lane) {
  const int n  = n0 + (lane & 15);
  const int kh = (lane >> 4) * 8;
  const _Float16* p = W + (size_t)n * ldw + k0 + kh;
  v8h lo = *(const v8h*)p;
  v8h hi = *(const v8h*)(p + 16);
  v16h r;
#pragma unroll
  for (int i = 0; i < 8; ++i) { r[i] = lo[i]; r[i + 8] = hi[i]; }
  return r;
}

// ---- Utility kernels ----

__global__ void cvt_f32_f16(const float* __restrict__ src, _Float16* __restrict__ dst, int n) {
  int i = blockIdx.x * blockDim.x + threadIdx.x;
  if (i < n) dst[i] = (_Float16)src[i];
}

__global__ void zero_h16(_Float16* __restrict__ p, int n) {
  int i = blockIdx.x * blockDim.x + threadIdx.x;
  if (i < n) p[i] = (_Float16)0.0f;
}

__global__ void zero_u32(unsigned* __restrict__ p) { *p = 0u; }

// ---- Prologue GEMM: pre0 = x @ w_ih0.T + b_ih0 + b_hh0 ----
// M=32768, N=1024, K=512.  32x64 tile/wave (8 wmma per K-step), explicit
// double-buffering: iteration k's WMMAs cover iteration k+1's loads + cvt.
__global__ __launch_bounds__(256)
void gemm_pre0(const float* __restrict__ x,        // [MS_][512] f32
               const _Float16* __restrict__ wih0,  // [1024][512] f16
               const float* __restrict__ b_ih0,
               const float* __restrict__ b_hh0,
               float* __restrict__ pre0)           // [MS_][1024] f32
{
  const int wave = blockIdx.x * 8 + (threadIdx.x >> 5);
  const int lane = threadIdx.x & 31;
  const int tm = wave >> 4;           // 0..1023 (32-row tiles)
  const int tn = wave & 15;           // 0..15   (64-wide strips)
  const int m0 = tm * 32, n0 = tn * 64;

  v8f c[8] = {};
  v16h a0 = load_a_f(x, DIN_, m0,      0, lane);
  v16h a1 = load_a_f(x, DIN_, m0 + 16, 0, lane);
  v16h b[4];
#pragma unroll
  for (int j = 0; j < 4; ++j) b[j] = load_b_w(wih0, DIN_, n0 + j * 16, 0, lane);

  for (int k0 = 32; k0 < DIN_; k0 += 32) {
    v16h na0 = load_a_f(x, DIN_, m0,      k0, lane);
    v16h na1 = load_a_f(x, DIN_, m0 + 16, k0, lane);
    v16h nb[4];
#pragma unroll
    for (int j = 0; j < 4; ++j) nb[j] = load_b_w(wih0, DIN_, n0 + j * 16, k0, lane);
#pragma unroll
    for (int j = 0; j < 4; ++j) {
      c[j]     = wmma16(a0, b[j], c[j]);
      c[4 + j] = wmma16(a1, b[j], c[4 + j]);
    }
    a0 = na0; a1 = na1;
#pragma unroll
    for (int j = 0; j < 4; ++j) b[j] = nb[j];
  }
#pragma unroll
  for (int j = 0; j < 4; ++j) {
    c[j]     = wmma16(a0, b[j], c[j]);
    c[4 + j] = wmma16(a1, b[j], c[4 + j]);
  }

#pragma unroll
  for (int r = 0; r < 2; ++r) {
    const int mr = m0 + r * 16 + (lane >> 4) * 8;
#pragma unroll
    for (int j = 0; j < 4; ++j) {
      const int n = n0 + j * 16 + (lane & 15);
      const float bias = b_ih0[n] + b_hh0[n];
#pragma unroll
      for (int i = 0; i < 8; ++i)
        pre0[(size_t)(mr + i) * H_ + n] = c[r * 4 + j][i] + bias;
    }
  }
}

// ---- Persistent recurrent kernel ----
// One launch runs all S+1 pipeline steps with a grid-wide barrier per step.
// blocks 0..15  : layer0 of step s   (skip at s == S)
// blocks 16..31 : layer1 of step s-1 (skip at s == 0)
// Each wave owns a fixed 16x32 output tile. Hidden states double-buffered by parity.
__global__ __launch_bounds__(256)
void rnn_persistent(const _Float16* __restrict__ whh0,
                    const _Float16* __restrict__ wih1,
                    const _Float16* __restrict__ whh1,
                    const float* __restrict__ pre0,
                    const float* __restrict__ b_ih1,
                    const float* __restrict__ b_hh1,
                    _Float16* __restrict__ h0buf,   // 2 x [64][1024]
                    _Float16* __restrict__ h1buf,   // 2 x [64][1024]
                    _Float16* __restrict__ rnn1,    // [B][S][H] f16
                    unsigned* __restrict__ bar)     // monotonic arrival counter
{
  const int phase = blockIdx.x >> 4;     // 0 or 1
  const int lb    = blockIdx.x & 15;
  const int wave  = threadIdx.x >> 5;
  const int lane  = threadIdx.x & 31;
  const int tile  = lb * 8 + wave;       // 0..127
  const int tm    = tile >> 5;           // 0..3  (M = 64)
  const int tn2   = tile & 31;           // 0..31 (N strips of 32)
  const int m0 = tm * 16, n0 = tn2 * 32;
  const int mr = m0 + (lane >> 4) * 8;
  const int nc = lane & 15;

  for (int s = 0; s <= S_; ++s) {
    if (phase == 0) {
      if (s < S_) {
        // h0_new = tanh(pre0[:,s,:] + h0_prev @ whh0.T)  -- pre0 already has both biases
        const _Float16* h0prev = h0buf + (size_t)((s + 1) & 1) * (B_ * H_);
        v8f c0 = {}, c1 = {};
        v16h a  = load_a_h(h0prev, H_, m0, 0, lane);
        v16h b0 = load_b_w(whh0, H_, n0,      0, lane);
        v16h b1 = load_b_w(whh0, H_, n0 + 16, 0, lane);
        for (int k0 = 32; k0 < H_; k0 += 32) {
          v16h an  = load_a_h(h0prev, H_, m0, k0, lane);
          v16h bn0 = load_b_w(whh0, H_, n0,      k0, lane);
          v16h bn1 = load_b_w(whh0, H_, n0 + 16, k0, lane);
          c0 = wmma16(a, b0, c0);
          c1 = wmma16(a, b1, c1);
          a = an; b0 = bn0; b1 = bn1;
        }
        c0 = wmma16(a, b0, c0);
        c1 = wmma16(a, b1, c1);

        _Float16* h0out = h0buf + (size_t)(s & 1) * (B_ * H_);
        v8f cc[2] = {c0, c1};
#pragma unroll
        for (int j = 0; j < 2; ++j) {
          const int n = n0 + j * 16 + nc;
#pragma unroll
          for (int i = 0; i < 8; ++i) {
            const int bb = mr + i;  // batch row
            float v = cc[j][i] + pre0[((size_t)bb * S_ + s) * H_ + n];
            h0out[bb * H_ + n] = (_Float16)tanhf(v);
          }
        }
      }
    } else {
      if (s > 0) {
        // t = s-1: h1_new = tanh(h0n[t] @ wih1.T + b_ih1 + h1_prev @ whh1.T + b_hh1)
        const int t = s - 1;
        const _Float16* h0cur  = h0buf + (size_t)(t & 1) * (B_ * H_);
        const _Float16* h1prev = h1buf + (size_t)((t + 1) & 1) * (B_ * H_);
        v8f c0 = {}, c1 = {};
        {
          v16h a  = load_a_h(h0cur, H_, m0, 0, lane);
          v16h b0 = load_b_w(wih1, H_, n0,      0, lane);
          v16h b1 = load_b_w(wih1, H_, n0 + 16, 0, lane);
          for (int k0 = 32; k0 < H_; k0 += 32) {
            v16h an  = load_a_h(h0cur, H_, m0, k0, lane);
            v16h bn0 = load_b_w(wih1, H_, n0,      k0, lane);
            v16h bn1 = load_b_w(wih1, H_, n0 + 16, k0, lane);
            c0 = wmma16(a, b0, c0);
            c1 = wmma16(a, b1, c1);
            a = an; b0 = bn0; b1 = bn1;
          }
          c0 = wmma16(a, b0, c0);
          c1 = wmma16(a, b1, c1);
        }
        {
          v16h a  = load_a_h(h1prev, H_, m0, 0, lane);
          v16h b0 = load_b_w(whh1, H_, n0,      0, lane);
          v16h b1 = load_b_w(whh1, H_, n0 + 16, 0, lane);
          for (int k0 = 32; k0 < H_; k0 += 32) {
            v16h an  = load_a_h(h1prev, H_, m0, k0, lane);
            v16h bn0 = load_b_w(whh1, H_, n0,      k0, lane);
            v16h bn1 = load_b_w(whh1, H_, n0 + 16, k0, lane);
            c0 = wmma16(a, b0, c0);
            c1 = wmma16(a, b1, c1);
            a = an; b0 = bn0; b1 = bn1;
          }
          c0 = wmma16(a, b0, c0);
          c1 = wmma16(a, b1, c1);
        }

        _Float16* h1out = h1buf + (size_t)(t & 1) * (B_ * H_);
        v8f cc[2] = {c0, c1};
#pragma unroll
        for (int j = 0; j < 2; ++j) {
          const int n = n0 + j * 16 + nc;
          const float bias = b_ih1[n] + b_hh1[n];
#pragma unroll
          for (int i = 0; i < 8; ++i) {
            const int bb = mr + i;
            _Float16 hv = (_Float16)tanhf(cc[j][i] + bias);
            h1out[bb * H_ + n] = hv;
            rnn1[((size_t)bb * S_ + t) * H_ + n] = hv;
          }
        }
      }
    }

    // ---- grid-wide barrier (monotonic counter, one arrival per block) ----
    __threadfence();          // make this block's global stores device-visible
    __syncthreads();
    if (threadIdx.x == 0) {
      __hip_atomic_fetch_add(bar, 1u, __ATOMIC_RELEASE, __HIP_MEMORY_SCOPE_AGENT);
      const unsigned target = (unsigned)STEP_BLOCKS * (unsigned)(s + 1);
      while (__hip_atomic_load(bar, __ATOMIC_RELAXED, __HIP_MEMORY_SCOPE_AGENT) < target)
        __builtin_amdgcn_s_sleep(2);
      (void)__hip_atomic_load(bar, __ATOMIC_ACQUIRE, __HIP_MEMORY_SCOPE_AGENT);
    }
    __syncthreads();
  }
}

// ---- Epilogue GEMM: out = rnn1 @ w_out.T + b_out.  M=32768, N=1024, K=1024 ----
// Same 32x64 tiling + explicit double-buffering as gemm_pre0.
__global__ __launch_bounds__(256)
void gemm_out(const _Float16* __restrict__ rnn1,  // [MS_][1024] f16
              const _Float16* __restrict__ wout,  // [1024][1024] f16
              const float* __restrict__ b_out,
              float* __restrict__ out)            // [MS_][1024] f32
{
  const int wave = blockIdx.x * 8 + (threadIdx.x >> 5);
  const int lane = threadIdx.x & 31;
  const int tm = wave >> 4;
  const int tn = wave & 15;
  const int m0 = tm * 32, n0 = tn * 64;

  v8f c[8] = {};
  v16h a0 = load_a_h(rnn1, H_, m0,      0, lane);
  v16h a1 = load_a_h(rnn1, H_, m0 + 16, 0, lane);
  v16h b[4];
#pragma unroll
  for (int j = 0; j < 4; ++j) b[j] = load_b_w(wout, H_, n0 + j * 16, 0, lane);

  for (int k0 = 32; k0 < H_; k0 += 32) {
    v16h na0 = load_a_h(rnn1, H_, m0,      k0, lane);
    v16h na1 = load_a_h(rnn1, H_, m0 + 16, k0, lane);
    v16h nb[4];
#pragma unroll
    for (int j = 0; j < 4; ++j) nb[j] = load_b_w(wout, H_, n0 + j * 16, k0, lane);
#pragma unroll
    for (int j = 0; j < 4; ++j) {
      c[j]     = wmma16(a0, b[j], c[j]);
      c[4 + j] = wmma16(a1, b[j], c[4 + j]);
    }
    a0 = na0; a1 = na1;
#pragma unroll
    for (int j = 0; j < 4; ++j) b[j] = nb[j];
  }
#pragma unroll
  for (int j = 0; j < 4; ++j) {
    c[j]     = wmma16(a0, b[j], c[j]);
    c[4 + j] = wmma16(a1, b[j], c[4 + j]);
  }

#pragma unroll
  for (int r = 0; r < 2; ++r) {
    const int mr = m0 + r * 16 + (lane >> 4) * 8;
#pragma unroll
    for (int j = 0; j < 4; ++j) {
      const int n = n0 + j * 16 + (lane & 15);
      const float bias = b_out[n];
#pragma unroll
      for (int i = 0; i < 8; ++i)
        out[(size_t)(mr + i) * DOUT_ + n] = c[r * 4 + j][i] + bias;
    }
  }
}

extern "C" void kernel_launch(void* const* d_in, const int* in_sizes, int n_in,
                              void* d_out, int out_size, void* d_ws, size_t ws_size,
                              hipStream_t stream) {
  const float* x     = (const float*)d_in[0];
  const float* w_ih0 = (const float*)d_in[1];
  const float* w_hh0 = (const float*)d_in[2];
  const float* b_ih0 = (const float*)d_in[3];
  const float* b_hh0 = (const float*)d_in[4];
  const float* w_ih1 = (const float*)d_in[5];
  const float* w_hh1 = (const float*)d_in[6];
  const float* b_ih1 = (const float*)d_in[7];
  const float* b_hh1 = (const float*)d_in[8];
  const float* w_out = (const float*)d_in[9];
  const float* b_out = (const float*)d_in[10];
  float* out = (float*)d_out;

  char* ws = (char*)d_ws;
  size_t off = 0;
  _Float16* wih0_h = (_Float16*)(ws + off); off += (size_t)H_ * DIN_ * 2;   // 1 MB
  _Float16* whh0_h = (_Float16*)(ws + off); off += (size_t)H_ * H_ * 2;     // 2 MB
  _Float16* wih1_h = (_Float16*)(ws + off); off += (size_t)H_ * H_ * 2;
  _Float16* whh1_h = (_Float16*)(ws + off); off += (size_t)H_ * H_ * 2;
  _Float16* wout_h = (_Float16*)(ws + off); off += (size_t)H_ * H_ * 2;
  _Float16* h0buf  = (_Float16*)(ws + off); off += (size_t)2 * B_ * H_ * 2; // double buffered
  _Float16* h1buf  = (_Float16*)(ws + off); off += (size_t)2 * B_ * H_ * 2;
  unsigned* bar    = (unsigned*)(ws + off); off += 256;                     // barrier counter
  _Float16* rnn1   = (_Float16*)(ws + off); off += (size_t)MS_ * H_ * 2;    // 67 MB
  float*    pre0   = (float*)   (ws + off); off += (size_t)MS_ * H_ * 4;    // 134 MB

  // 1) Weight conversion f32 -> f16 (weights become L2-resident for all later kernels)
  cvt_f32_f16<<<(H_ * DIN_ + 255) / 256, 256, 0, stream>>>(w_ih0, wih0_h, H_ * DIN_);
  cvt_f32_f16<<<(H_ * H_  + 255) / 256, 256, 0, stream>>>(w_hh0, whh0_h, H_ * H_);
  cvt_f32_f16<<<(H_ * H_  + 255) / 256, 256, 0, stream>>>(w_ih1, wih1_h, H_ * H_);
  cvt_f32_f16<<<(H_ * H_  + 255) / 256, 256, 0, stream>>>(w_hh1, whh1_h, H_ * H_);
  cvt_f32_f16<<<(H_ * H_  + 255) / 256, 256, 0, stream>>>(w_out, wout_h, H_ * H_);

  // 2) Zero initial hidden states + barrier counter (graph replays self-initialize)
  zero_h16<<<(4 * B_ * H_ + 255) / 256, 256, 0, stream>>>(h0buf, 4 * B_ * H_);
  zero_u32<<<1, 1, 0, stream>>>(bar);

  // 3) Parallel prologue: pre0 = x @ w_ih0.T + b_ih0 + b_hh0
  gemm_pre0<<<2048, 256, 0, stream>>>(x, wih0_h, b_ih0, b_hh0, pre0);

  // 4) Entire recurrence in ONE persistent launch (grid barrier per pipeline step)
  rnn_persistent<<<STEP_BLOCKS, 256, 0, stream>>>(whh0_h, wih1_h, whh1_h, pre0,
                                                  b_ih1, b_hh1, h0buf, h1buf, rnn1, bar);

  // 5) Parallel epilogue: out = rnn1 @ w_out.T + b_out
  gemm_out<<<2048, 256, 0, stream>>>(rnn1, wout_h, b_out, out);
}
